// att_lstm_2405181686412
// MI455X (gfx1250) — matrix operand order repaired and checked by reading, not compiled, not run on hardware
//
#include <hip/hip_runtime.h>
#include <hip/hip_bf16.h>
#include <math.h>

// ---------------------------------------------------------------------------
// CDNA5 (gfx1250) attention-LSTM forward. All big GEMMs run on
// v_wmma_f32_16x16x32_bf16 with double-buffered LDS slabs filled by
// GLOBAL_LOAD_ASYNC_TO_LDS_B128 (ASYNCcnt) when the toolchain exposes it.
// wave32; 256-thread blocks = 8 waves.
// ---------------------------------------------------------------------------

typedef __attribute__((ext_vector_type(16))) __bf16 v16bf;
typedef __attribute__((ext_vector_type(8)))  float  v8f;
typedef __attribute__((ext_vector_type(4)))  int    v4i;

#define LDSTRIDE 40   // halfs per LDS row: 32 payload + 8 pad (keeps 16B align)

#if defined(__gfx1250__) && __has_builtin(__builtin_amdgcn_global_load_async_to_lds_b128) && \
    __has_builtin(__builtin_amdgcn_s_wait_asynccnt)
#define USE_ASYNC_LDS 1
#else
#define USE_ASYNC_LDS 0
#endif

__device__ __forceinline__ unsigned short f2bf(float f) {
  unsigned int u = __float_as_uint(f);
  u += 0x7fffu + ((u >> 16) & 1u);          // round-to-nearest-even
  return (unsigned short)(u >> 16);
}
__device__ __forceinline__ float bf2f(unsigned short h) {
  return __uint_as_float(((unsigned int)h) << 16);
}
__device__ __forceinline__ float sigf(float x) { return 1.0f / (1.0f + __expf(-x)); }

__device__ __forceinline__ void wait_async_lds() {
#if USE_ASYNC_LDS
  __builtin_amdgcn_s_wait_asynccnt(0);
#endif
}

// A fragment (16x32 bf16, MxK): lane m%16 holds row M. lanes<16: K in
// [0..7]U[16..23]; lanes>=16: K in [8..15]U[24..31]. (ISA 7.12.2)
__device__ __forceinline__ v16bf load_frag_a(const unsigned short* lds, int row, int lane) {
  const int kb = (lane < 16) ? 0 : 8;
  union { v16bf v; uint4 q[2]; } f;
  const unsigned short* p = lds + row * LDSTRIDE + kb;
  f.q[0] = *(const uint4*)(p);
  f.q[1] = *(const uint4*)(p + 16);
  return f.v;
}
// B fragment (32x16 bf16, KxN): lane holds column N=lane%16; lanes<16 K in
// [0..7]U[16..23]; lanes>=16: K in [8..15]U[24..31] (same as A).
__device__ __forceinline__ v16bf load_frag_b(const unsigned short* lds, int col, int lane) {
  const int kb = (lane < 16) ? 0 : 8;
  union { v16bf v; uint4 q[2]; } f;
  const unsigned short* p = lds + col * LDSTRIDE + kb;
  f.q[0] = *(const uint4*)(p);
  f.q[1] = *(const uint4*)(p + 16);
  return f.v;
}

// C = A(M,K) @ W(N,K)^T [+bias] [relu]. Workgroup tile 128x128, wave tile
// 64x32 (eight 16x16x32 WMMA accumulators). Double-buffered 32-wide K slabs.
// M,N multiples of 128; K multiple of 32; lda/ldb multiples of 8.
template <int HAS_BIAS, int RELU, int OUTBF>
__global__ __launch_bounds__(256) void gemm_wmma_kernel(
    const unsigned short* __restrict__ A,
    const unsigned short* __restrict__ W,
    const float* __restrict__ bias,
    void* __restrict__ Cout,
    int K, int lda, int ldb, int ldc) {
  __shared__ unsigned short lds_a[2][128 * LDSTRIDE];
  __shared__ unsigned short lds_b[2][128 * LDSTRIDE];

  const int t    = threadIdx.x;
  const int wave = t >> 5, lane = t & 31;
  const int wm   = wave >> 2;        // 0..1 -> 64-row slab
  const int wn   = wave & 3;         // 0..3 -> 32-col slab
  const int m0   = blockIdx.y * 128;
  const int n0   = blockIdx.x * 128;

  // Slab fill: 128 rows x 32 halfs = 512 x 16B chunks per matrix;
  // 256 threads x 2 iterations; unit u -> row u>>2, 16B chunk (u&3).
  auto fill = [&](int buf, int kt) {
#pragma unroll
    for (int i = 0; i < 2; ++i) {
      const int u = t + 256 * i;
      const int row = u >> 2, qo = (u & 3) * 8;      // half offset
      const unsigned short* ga = A + (size_t)(m0 + row) * lda + kt + qo;
      const unsigned short* gb = W + (size_t)(n0 + row) * ldb + kt + qo;
      unsigned short* da = &lds_a[buf][row * LDSTRIDE + qo];
      unsigned short* db = &lds_b[buf][row * LDSTRIDE + qo];
#if USE_ASYNC_LDS
      __builtin_amdgcn_global_load_async_to_lds_b128(
          (__attribute__((address_space(1))) v4i*)ga,
          (__attribute__((address_space(3))) v4i*)da, 0, 0);
      __builtin_amdgcn_global_load_async_to_lds_b128(
          (__attribute__((address_space(1))) v4i*)gb,
          (__attribute__((address_space(3))) v4i*)db, 0, 0);
#else
      *(uint4*)da = *(const uint4*)ga;
      *(uint4*)db = *(const uint4*)gb;
#endif
    }
  };

  v8f acc[4][2];
#pragma unroll
  for (int i = 0; i < 4; ++i)
#pragma unroll
    for (int j = 0; j < 2; ++j)
#pragma unroll
      for (int r = 0; r < 8; ++r) acc[i][j][r] = 0.0f;

  fill(0, 0);
  wait_async_lds();
  __syncthreads();

  int buf = 0;
  for (int kt = 0; kt < K; kt += 32) {
    const bool more = (kt + 32) < K;
    if (more) fill(buf ^ 1, kt + 32);   // prefetch next slab (other buffer)

    v16bf afrag[4], bfrag[2];
#pragma unroll
    for (int ms = 0; ms < 4; ++ms)
      afrag[ms] = load_frag_a(lds_a[buf], wm * 64 + ms * 16 + (lane & 15), lane);
#pragma unroll
    for (int ns = 0; ns < 2; ++ns)
      bfrag[ns] = load_frag_b(lds_b[buf], wn * 32 + ns * 16 + (lane & 15), lane);
#pragma unroll
    for (int ms = 0; ms < 4; ++ms)
#pragma unroll
      for (int ns = 0; ns < 2; ++ns)
        acc[ms][ns] = __builtin_amdgcn_wmma_f32_16x16x32_bf16(
            false, afrag[ms], false, bfrag[ns], (short)0, acc[ms][ns], false, false);

    wait_async_lds();
    __syncthreads();
    buf ^= 1;
  }

  // ---- epilogue: branchless, bias hoisted (col depends only on ns) ----
  const int lm = (lane < 16) ? 0 : 8;
  const int ln = lane & 15;
  float bv[2] = {0.f, 0.f};
  if constexpr (HAS_BIAS) {
#pragma unroll
    for (int ns = 0; ns < 2; ++ns) bv[ns] = bias[n0 + wn * 32 + ns * 16 + ln];
  }
#pragma unroll
  for (int ms = 0; ms < 4; ++ms)
#pragma unroll
    for (int ns = 0; ns < 2; ++ns) {
      const size_t base =
          (size_t)(m0 + wm * 64 + ms * 16 + lm) * ldc + (n0 + wn * 32 + ns * 16 + ln);
#pragma unroll
      for (int r = 0; r < 8; ++r) {
        float v = acc[ms][ns][r];
        if constexpr (HAS_BIAS) v += bv[ns];
        if constexpr (RELU) v = fmaxf(v, 0.0f);
        if constexpr (OUTBF)
          ((unsigned short*)Cout)[base + (size_t)r * ldc] = f2bf(v);
        else
          ((float*)Cout)[base + (size_t)r * ldc] = v;
      }
    }
}

// In-place LayerNorm over rows of length L (L multiple of 256, <=2048):
// x = (x-mean)*rsqrt(var+eps)*g + b (+ other[row*stride + off + j])
__global__ __launch_bounds__(256) void layernorm_kernel(
    float* __restrict__ x, const float* __restrict__ g, const float* __restrict__ b,
    const float* __restrict__ other, int L, int ld,
    long long other_row_stride, long long other_off) {
  __shared__ float s1[256], s2[256];
  const int row = blockIdx.x, t = threadIdx.x;
  float* xr = x + (size_t)row * ld;
  const int per = L / 256;
  float v[8];
  float a1 = 0.f, a2 = 0.f;
  for (int i = 0; i < per; ++i) {
    float f = xr[t + 256 * i];
    v[i] = f; a1 += f; a2 += f * f;
  }
  s1[t] = a1; s2[t] = a2; __syncthreads();
  for (int off = 128; off > 0; off >>= 1) {
    if (t < off) { s1[t] += s1[t + off]; s2[t] += s2[t + off]; }
    __syncthreads();
  }
  const float mean = s1[0] / (float)L;
  const float var  = s2[0] / (float)L - mean * mean;
  const float rs   = rsqrtf(var + 1e-5f);
  for (int i = 0; i < per; ++i) {
    const int j = t + 256 * i;
    float o = (v[i] - mean) * rs * g[j] + b[j];
    if (other) o += other[(size_t)row * other_row_stride + other_off + j];
    xr[j] = o;
  }
}

// LSTM pointwise cell + LayerNorm(c) fused. gates: (B,2048) = [i|f|g|o].
__global__ __launch_bounds__(256) void lstm_cell_kernel(
    const float* __restrict__ gates, float* __restrict__ c,
    const float* __restrict__ g_c, const float* __restrict__ b_c,
    unsigned short* __restrict__ h_bf, unsigned short* __restrict__ lstm_bf, int tstep) {
  __shared__ float s1[256], s2[256];
  const int b = blockIdx.x, t = threadIdx.x;
  const float* gr = gates + (size_t)b * 2048;
  float cn[2];
  float a1 = 0.f, a2 = 0.f;
#pragma unroll
  for (int i = 0; i < 2; ++i) {
    const int j = t + 256 * i;
    const float ig = sigf(gr[j]);
    const float fg = sigf(gr[512 + j]);
    const float gg = tanhf(gr[1024 + j]);
    const float cv = fg * c[(size_t)b * 512 + j] + ig * gg;
    cn[i] = cv; a1 += cv; a2 += cv * cv;
  }
  s1[t] = a1; s2[t] = a2; __syncthreads();
  for (int off = 128; off > 0; off >>= 1) {
    if (t < off) { s1[t] += s1[t + off]; s2[t] += s2[t + off]; }
    __syncthreads();
  }
  const float mean = s1[0] / 512.f;
  const float var  = s2[0] / 512.f - mean * mean;
  const float rs   = rsqrtf(var + 1e-5f);
#pragma unroll
  for (int i = 0; i < 2; ++i) {
    const int j = t + 256 * i;
    const float og = sigf(gr[1536 + j]);
    const float hv = og * tanhf((cn[i] - mean) * rs * g_c[j] + b_c[j]);
    c[(size_t)b * 512 + j] = cn[i];
    const unsigned short hb = f2bf(hv);
    h_bf[(size_t)b * 512 + j] = hb;
    lstm_bf[((size_t)b * 16 + tstep) * 512 + j] = hb;
  }
}

// scores -> softmax over SEQ=16 -> u[b,h,:] = sum_s attn[s]*lstm_out[b,s,:]
__global__ __launch_bounds__(128) void attn_kernel(
    const float* __restrict__ q, const float* __restrict__ kmat,
    const unsigned short* __restrict__ lstm_bf, unsigned short* __restrict__ u_bf) {
  __shared__ float sc[16];
  const int bh = blockIdx.x, b = bh >> 2, h = bh & 3;
  const int t = threadIdx.x;
  const int s = t >> 3, l8 = t & 7;   // 8 lanes per timestep
  const float* qr = q + (size_t)b * 512 + h * 128;
  const float* kr = kmat + ((size_t)b * 16 + s) * 512 + h * 128;
  float p = 0.f;
#pragma unroll
  for (int j = 0; j < 16; ++j) p += kr[l8 + 8 * j] * qr[l8 + 8 * j];
  p += __shfl_xor(p, 1, 32); p += __shfl_xor(p, 2, 32); p += __shfl_xor(p, 4, 32);
  if (l8 == 0) sc[s] = p * 0.08838834764831845f;   // 1/sqrt(128)
  __syncthreads();
  float mx = -1e30f;
#pragma unroll
  for (int i = 0; i < 16; ++i) mx = fmaxf(mx, sc[i]);
  float w[16]; float den = 0.f;
#pragma unroll
  for (int i = 0; i < 16; ++i) { w[i] = __expf(sc[i] - mx); den += w[i]; }
  const float inv = 1.0f / den;
  const unsigned short* lb = lstm_bf + (size_t)b * 16 * 512;
#pragma unroll
  for (int i = 0; i < 4; ++i) {
    const int d = t + 128 * i;
    float acc = 0.f;
#pragma unroll
    for (int ss = 0; ss < 16; ++ss) acc += (w[ss] * inv) * bf2f(lb[ss * 512 + d]);
    u_bf[((size_t)b * 4 + h) * 512 + d] = f2bf(acc);
  }
}

__global__ __launch_bounds__(128) void outdot_kernel(
    const float* __restrict__ out_h, const float* __restrict__ out_W,
    const float* __restrict__ out_b, float* __restrict__ d_out) {
  __shared__ float s[128];
  const int b = blockIdx.x, t = threadIdx.x;
  float p = 0.f;
#pragma unroll
  for (int i = 0; i < 4; ++i) { const int j = t + 128 * i; p += out_h[(size_t)b * 512 + j] * out_W[j]; }
  s[t] = p; __syncthreads();
  for (int off = 64; off > 0; off >>= 1) { if (t < off) s[t] += s[t + off]; __syncthreads(); }
  if (t == 0) d_out[1 + b] = s[0] + out_b[0];
}

__global__ __launch_bounds__(256) void loss_kernel(
    const float* __restrict__ outv, const float* __restrict__ label, float* __restrict__ loss) {
  __shared__ float s[256];
  const int t = threadIdx.x;
  float a = 0.f;
  for (int b = t; b < 2048; b += 256) {
    const float d = outv[b] - label[(size_t)b * 16 + 15];
    a += d * d;
  }
  s[t] = a; __syncthreads();
  for (int off = 128; off > 0; off >>= 1) { if (t < off) s[t] += s[t + off]; __syncthreads(); }
  if (t == 0) *loss = s[0] / 2048.0f;
}

// vectorized fp32 -> bf16 (n multiple of 4)
__global__ void cvt_bf16_kernel(const float* __restrict__ in, unsigned short* __restrict__ out, int n4) {
  const int i = blockIdx.x * 256 + threadIdx.x;
  if (i < n4) {
    const float4 f = *(const float4*)(in + (size_t)i * 4);
    ushort4 o;
    o.x = f2bf(f.x); o.y = f2bf(f.y); o.z = f2bf(f.z); o.w = f2bf(f.w);
    *(ushort4*)(out + (size_t)i * 4) = o;
  }
}

// fp32 (rows,1024) -> bf16 strided into concat (rows,4096) at column offset
// baked into `out`. One launch per modality tensor.
__global__ void cvt_concat_kernel(const float* __restrict__ in, unsigned short* __restrict__ out) {
  const int i4 = (blockIdx.x * 256 + threadIdx.x) * 4;
  const float4 f = *(const float4*)(in + i4);
  ushort4 o;
  o.x = f2bf(f.x); o.y = f2bf(f.y); o.z = f2bf(f.z); o.w = f2bf(f.w);
  const int row = i4 >> 10, col = i4 & 1023;
  *(ushort4*)(out + (size_t)row * 4096 + col) = o;
}

// ---------------------------------------------------------------------------
extern "C" void kernel_launch(void* const* d_in, const int* in_sizes, int n_in,
                              void* d_out, int out_size, void* d_ws, size_t ws_size,
                              hipStream_t stream) {
  (void)in_sizes; (void)n_in; (void)out_size; (void)ws_size;
  constexpr int B = 2048, SEQ = 16, DIM = 1024, HID = 512;
  constexpr int BS = B * SEQ;            // 32768
  constexpr int G = 4 * HID;             // 2048 gate width

  const float* visual = (const float*)d_in[0];
  const float* text   = (const float*)d_in[1];
  const float* user   = (const float*)d_in[2];
  const float* cat    = (const float*)d_in[3];
  const float* label  = (const float*)d_in[4];
  const float* h0     = (const float*)d_in[5];
  const float* c0     = (const float*)d_in[6];
  const float* fus_W  = (const float*)d_in[7];
  const float* fus_b  = (const float*)d_in[8];
  const float* W_ih   = (const float*)d_in[9];
  const float* W_hh   = (const float*)d_in[10];
  const float* g_ih   = (const float*)d_in[11];
  const float* b_ih   = (const float*)d_in[12];
  const float* g_hh   = (const float*)d_in[13];
  const float* b_hh   = (const float*)d_in[14];
  const float* g_c    = (const float*)d_in[15];
  const float* b_c    = (const float*)d_in[16];
  const float* wq_W   = (const float*)d_in[17];
  const float* wq_b   = (const float*)d_in[18];
  const float* wk_W   = (const float*)d_in[19];
  const float* wk_b   = (const float*)d_in[20];
  const float* wv_W   = (const float*)d_in[21];
  const float* wv_b   = (const float*)d_in[22];
  const float* wout_W = (const float*)d_in[23];
  const float* wout_b = (const float*)d_in[24];
  const float* out_W  = (const float*)d_in[25];
  const float* out_b  = (const float*)d_in[26];

  // ---- carve workspace ----
  char* wp = (char*)d_ws;
  auto alloc = [&](size_t bytes) -> void* {
    void* r = (void*)wp; wp += (bytes + 255) & ~(size_t)255; return r;
  };
  unsigned short* fusW_bf = (unsigned short*)alloc((size_t)DIM * 4 * DIM * 2);
  unsigned short* Wih_bf  = (unsigned short*)alloc((size_t)G * DIM * 2);
  unsigned short* Whh_bf  = (unsigned short*)alloc((size_t)G * HID * 2);
  unsigned short* wq_bf   = (unsigned short*)alloc((size_t)512 * 512 * 2);
  unsigned short* wk_bf   = (unsigned short*)alloc((size_t)512 * 512 * 2);
  unsigned short* wv_bf   = (unsigned short*)alloc((size_t)4 * 512 * 512 * 2);
  unsigned short* wout_bf = (unsigned short*)alloc((size_t)512 * 2048 * 2);
  unsigned short* mods_bf = (unsigned short*)alloc((size_t)BS * 4 * DIM * 2); // 256 MB
  unsigned short* vt_bf   = (unsigned short*)alloc((size_t)BS * DIM * 2);     // 64 MB
  float*          Xg      = (float*)alloc((size_t)BS * G * 4);                // 256 MB
  float*          Hg      = (float*)alloc((size_t)B * G * 4);                 // 16 MB
  unsigned short* h_bf    = (unsigned short*)alloc((size_t)B * HID * 2);
  float*          cbuf    = (float*)alloc((size_t)B * HID * 4);
  unsigned short* lstm_bf = (unsigned short*)alloc((size_t)B * SEQ * HID * 2);
  float*          qbuf    = (float*)alloc((size_t)B * 512 * 4);
  float*          kbuf    = (float*)alloc((size_t)BS * 512 * 4);              // 64 MB
  unsigned short* u_bf    = (unsigned short*)alloc((size_t)B * 2048 * 2);
  unsigned short* ctx_bf  = (unsigned short*)alloc((size_t)B * 2048 * 2);
  float*          outh    = (float*)alloc((size_t)B * 512 * 4);

  auto cvt = [&](const float* in, unsigned short* out, int n) {
    cvt_bf16_kernel<<<(n / 4 + 255) / 256, 256, 0, stream>>>(in, out, n / 4);
  };
  cvt(fus_W, fusW_bf, DIM * 4 * DIM);
  cvt(W_ih,  Wih_bf,  G * DIM);
  cvt(W_hh,  Whh_bf,  G * HID);
  cvt(wq_W,  wq_bf,   512 * 512);
  cvt(wk_W,  wk_bf,   512 * 512);
  cvt(wv_W,  wv_bf,   4 * 512 * 512);
  cvt(wout_W, wout_bf, 512 * 2048);
  cvt(h0, h_bf, B * HID);
  (void)hipMemcpyAsync(cbuf, c0, (size_t)B * HID * 4, hipMemcpyDeviceToDevice, stream);

  // concat modality tensors -> bf16 (32768 x 4096)
  {
    const int blocks = (BS * DIM / 4) / 256;   // 32768
    cvt_concat_kernel<<<blocks, 256, 0, stream>>>(visual, mods_bf + 0 * DIM);
    cvt_concat_kernel<<<blocks, 256, 0, stream>>>(text,   mods_bf + 1 * DIM);
    cvt_concat_kernel<<<blocks, 256, 0, stream>>>(user,   mods_bf + 2 * DIM);
    cvt_concat_kernel<<<blocks, 256, 0, stream>>>(cat,    mods_bf + 3 * DIM);
  }

  // fusion: v_t = relu(mods @ fus_W^T + b) -> bf16 (32768 x 1024)
  gemm_wmma_kernel<1, 1, 1><<<dim3(DIM / 128, BS / 128), 256, 0, stream>>>(
      mods_bf, fusW_bf, fus_b, vt_bf, 4 * DIM, 4 * DIM, 4 * DIM, DIM);

  // Xg = v_t @ W_ih^T (fp32), then in-place LN with g_ih/b_ih (all timesteps)
  gemm_wmma_kernel<0, 0, 0><<<dim3(G / 128, BS / 128), 256, 0, stream>>>(
      vt_bf, Wih_bf, nullptr, Xg, DIM, DIM, DIM, G);
  layernorm_kernel<<<BS, 256, 0, stream>>>(Xg, g_ih, b_ih, nullptr, G, G, 0, 0);

  // recurrent loop
  for (int t = 0; t < SEQ; ++t) {
    gemm_wmma_kernel<0, 0, 0><<<dim3(G / 128, B / 128), 256, 0, stream>>>(
        h_bf, Whh_bf, nullptr, Hg, HID, HID, HID, G);
    layernorm_kernel<<<B, 256, 0, stream>>>(
        Hg, g_hh, b_hh, Xg, G, G, (long long)SEQ * G, (long long)t * G);
    lstm_cell_kernel<<<B, 256, 0, stream>>>(Hg, cbuf, g_c, b_c, h_bf, lstm_bf, t);
  }

  // q = hn @ wq^T + b ; k = lstm_out @ wk^T + b
  gemm_wmma_kernel<1, 0, 0><<<dim3(512 / 128, B / 128), 256, 0, stream>>>(
      h_bf, wq_bf, wq_b, qbuf, HID, HID, HID, 512);
  gemm_wmma_kernel<1, 0, 0><<<dim3(512 / 128, BS / 128), 256, 0, stream>>>(
      lstm_bf, wk_bf, wk_b, kbuf, HID, HID, HID, 512);

  // softmax + attn-weighted sum -> u (bf16)
  attn_kernel<<<B * 4, 128, 0, stream>>>(qbuf, kbuf, lstm_bf, u_bf);

  // ctx[:, h*512:(h+1)*512] = u[:,h,:] @ wv_W[h]^T + wv_b[h]  (bf16 out)
  for (int h = 0; h < 4; ++h) {
    gemm_wmma_kernel<1, 0, 1><<<dim3(512 / 128, B / 128), 256, 0, stream>>>(
        u_bf + h * 512, wv_bf + (size_t)h * 512 * 512, wv_b + h * 512, ctx_bf + h * 512,
        HID, 2048, 512, 2048);
  }

  // out_h = ctx @ wout^T + b
  gemm_wmma_kernel<1, 0, 0><<<dim3(512 / 128, B / 128), 256, 0, stream>>>(
      ctx_bf, wout_bf, wout_b, outh, 2048, 2048, 2048, 512);

  // output + loss -> d_out[0]=loss, d_out[1..2048]=output
  outdot_kernel<<<B, 128, 0, stream>>>(outh, out_W, out_b, (float*)d_out);
  loss_kernel<<<1, 256, 0, stream>>>((const float*)d_out + 1, label, (float*)d_out);
}
